// MultiHeadAttention_5342939316866
// MI455X (gfx1250) — matrix operand order repaired and checked by reading, not compiled, not run on hardware
//
#include <hip/hip_runtime.h>
#include <hip/hip_bf16.h>

// ---------------------------------------------------------------------------
// MHA forward for gfx1250 (MI455X): all matmuls via v_wmma_f32_16x16x32_f16.
// Compute-bound (~137 GFLOP vs ~8us of HBM traffic), so: f16 data / f32
// accumulate, fused flash-attention so the S x S score matrix never leaves
// registers/LDS, contiguous fragment loads per the CDNA5 WMMA VGPR layout
// tables (ISA 7.12.2), and double-buffered K/V tile staging through the
// CDNA5 async global->LDS path (GLOBAL_LOAD_ASYNC_TO_LDS_B128 / ASYNCcnt).
// ---------------------------------------------------------------------------

typedef _Float16 f16;
typedef __attribute__((ext_vector_type(16))) _Float16 v16h;
typedef __attribute__((ext_vector_type(8)))  _Float16 h8;
typedef __attribute__((ext_vector_type(8)))  float    v8f;

#define BATCH   4
#define SEQ     2048
#define DMODEL  1024
#define NHEADS  16
#define HEADD   64
#define BS      (BATCH * SEQ)          // 8192 rows

union V16 { v16h v; h8 h[2]; };

__device__ __forceinline__ v8f wmma_f16(v16h a, v16h b, v8f c) {
  // (neg_a, A, neg_b, B, c_mod, C, reuse_a, reuse_b)
  return __builtin_amdgcn_wmma_f32_16x16x32_f16(false, a, false, b, (short)0, c,
                                                false, false);
}

// A-fragment, 16x32 f16, row-major source with leading dim `ld` (halves).
// ISA layout: lanes 0-15 -> M=lane, halves = K {k0..k0+7, k0+16..k0+23};
//             lanes 16-31 -> M=lane-16, halves = K {k0+8..15, k0+24..31}.
__device__ __forceinline__ v16h load_a16x32(const f16* base, int ld, int m0, int k0) {
  const int l   = threadIdx.x & 31;
  const int row = m0 + (l & 15);
  const int kb  = k0 + ((l >> 4) << 3);
  const f16* p  = base + (size_t)row * ld + kb;
  V16 u;
  u.h[0] = *(const h8*)(p);
  u.h[1] = *(const h8*)(p + 16);
  return u.v;
}

// B-fragment, 32x16 f16, row-major [K][N] source with leading dim `ld`.
// ISA layout: lanes 0-15 -> N=n0+lane, halves = K {k0..k0+7, k0+16..k0+23};
//             lanes 16-31 -> N=n0+lane-16, halves = K {k0+8..15, k0+24..31}.
__device__ __forceinline__ v16h load_b32x16(const f16* base, int ld, int k0, int n0) {
  const int l  = threadIdx.x & 31;
  const int kb = k0 + ((l >> 4) << 3);
  const f16* p = base + (size_t)kb * ld + n0 + (l & 15);
  v16h v;
#pragma unroll
  for (int s = 0; s < 8; ++s) {
    v[s]     = p[(size_t)s * ld];
    v[8 + s] = p[(size_t)(16 + s) * ld];
  }
  return v;
}

// CDNA5 async global->LDS copy of 16 bytes (per lane). VDST holds the LDS
// byte address, VADDR the 64-bit global address (GV mode). Tracked by
// ASYNCcnt; drained with s_wait_asynccnt before the consuming barrier.
__device__ __forceinline__ void async_ld16(unsigned ldsOff, const f16* gptr) {
  unsigned long long ga = (unsigned long long)(const void*)gptr;
  asm volatile("global_load_async_to_lds_b128 %0, %1, off"
               :: "v"(ldsOff), "v"(ga)
               : "memory");
}

__device__ __forceinline__ void wait_async0() {
  asm volatile("s_wait_asynccnt 0x0" ::: "memory");
}

// ---------------------------------------------------------------------------
// f32 -> f16 conversions
// ---------------------------------------------------------------------------
__global__ void cvt_f32_f16(const float* __restrict__ in, f16* __restrict__ out, int n) {
  int i = blockIdx.x * blockDim.x + threadIdx.x;
  if (i < n) out[i] = (f16)in[i];
}

// Wo is used as out @ Wo^T; store WoT[k][n] = Wo[n][k] so B-frag rows are
// contiguous in the output-projection GEMM.
__global__ void cvt_wo_transpose(const float* __restrict__ in, f16* __restrict__ out) {
  int i = blockIdx.x * blockDim.x + threadIdx.x;   // over 1024*1024
  int k = i >> 10, n = i & 1023;
  out[(size_t)k * DMODEL + n] = (f16)in[(size_t)n * DMODEL + k];
}

// ---------------------------------------------------------------------------
// QKV projection: per head h, [8192 x 1024] x [1024 x 64] (+bias) -> f16
// stored as [B,H,S,64]. Block = 8 waves = 2 row-tiles x 4 col-tiles
// (32 rows x 64 cols). Two accumulators per wave to break WMMA RAW chains.
// ---------------------------------------------------------------------------
__global__ void __launch_bounds__(256)
qkv_gemm(const f16* __restrict__ xh,
         const f16* __restrict__ Wqh, const f16* __restrict__ Wkh, const f16* __restrict__ Wvh,
         const float* __restrict__ bq, const float* __restrict__ bk, const float* __restrict__ bv,
         f16* __restrict__ Qh, f16* __restrict__ Kh, f16* __restrict__ Vh) {
  const int which = blockIdx.z;
  const f16*   W    = which == 0 ? Wqh : (which == 1 ? Wkh : Wvh);
  const float* bias = which == 0 ? bq  : (which == 1 ? bk  : bv);
  f16*         out  = which == 0 ? Qh  : (which == 1 ? Kh  : Vh);

  const int h   = blockIdx.y;
  const int tid = threadIdx.x;
  const int w   = tid >> 5, l = tid & 31;
  const int m0  = blockIdx.x * 32 + (w >> 2) * 16;
  const int n0  = (w & 3) * 16;
  const f16* Wh = W + (size_t)h * DMODEL * HEADD;

  v8f c0 = {}, c1 = {};
  for (int k0 = 0; k0 < DMODEL; k0 += 64) {
    // pull the A stream ahead into cache (global_prefetch_b8)
    if (k0 + 256 < DMODEL)
      __builtin_prefetch(xh + (size_t)(m0 + (l & 15)) * DMODEL + k0 + 256, 0, 1);
    v16h a0 = load_a16x32(xh, DMODEL, m0, k0);
    v16h b0 = load_b32x16(Wh, HEADD, k0, n0);
    c0 = wmma_f16(a0, b0, c0);
    v16h a1 = load_a16x32(xh, DMODEL, m0, k0 + 32);
    v16h b1 = load_b32x16(Wh, HEADD, k0 + 32, n0);
    c1 = wmma_f16(a1, b1, c1);
  }
  v8f c = c0 + c1;

  const float bb     = bias[h * HEADD + n0 + (l & 15)];
  const int   rowAdd = (l >> 4) * 8;
#pragma unroll
  for (int r = 0; r < 8; ++r) {
    int grow = m0 + rowAdd + r;            // global row in [0, 8192)
    int b    = grow >> 11;                 // /2048
    int s    = grow & 2047;
    out[(((size_t)(b * NHEADS + h)) * SEQ + s) * HEADD + n0 + (l & 15)] =
        (f16)(c[r] + bb);
  }
}

// ---------------------------------------------------------------------------
// Fused flash attention per (b,h). Block = 4 waves, 64 q rows; each wave
// owns 16 q rows. Key loop in blocks of 32 with DOUBLE-BUFFERED K/V tiles
// staged by async global->LDS copies (overlap next tile fetch with WMMAs):
//   S^T = K_tile(16x64) * Q^T(64x16)  -> WMMA, operands contiguous from LDS
//   online softmax (wave32 shfl_xor(16) reductions)
//   P^T staged transposed to per-wave LDS -> A-frag for O += P * V (WMMA)
// ---------------------------------------------------------------------------
__device__ __forceinline__ void stage_kv_async(const f16* Kp, const f16* Vp, int kb,
                                               f16* KtBuf, f16* VtBuf, int tid) {
#pragma unroll
  for (int i = 0; i < 2; ++i) {
    int j   = tid + i * 128;          // chunk id 0..255 (16B chunks)
    int key = j >> 3, dc = (j & 7) * 8;
    const f16* gk = Kp + (size_t)(kb + key) * HEADD + dc;
    const f16* gv = Vp + (size_t)(kb + key) * HEADD + dc;
    async_ld16((unsigned)(size_t)(KtBuf + key * 64 + dc), gk);
    async_ld16((unsigned)(size_t)(VtBuf + key * 64 + dc), gv);
  }
}

__global__ void __launch_bounds__(128)
flash_attn(const f16* __restrict__ Qh, const f16* __restrict__ Kh,
           const f16* __restrict__ Vh, f16* __restrict__ Ah) {
  __shared__ __attribute__((aligned(64))) f16 qT[64 * 64];          // [d][q_local]
  __shared__ __attribute__((aligned(64))) f16 Kt[2][32 * 64];       // double buffer
  __shared__ __attribute__((aligned(64))) f16 Vt[2][32 * 64];       // double buffer
  __shared__ __attribute__((aligned(64))) f16 pBuf[4][16 * 32];     // per wave [q][key]

  const int b = blockIdx.z, h = blockIdx.y;
  const int q0blk = blockIdx.x * 64;
  const int tid = threadIdx.x;
  const int w = tid >> 5, l = tid & 31;
  const size_t headBase = ((size_t)(b * NHEADS + h)) * SEQ * HEADD;
  const f16* Qp = Qh + headBase;
  const f16* Kp = Kh + headBase;
  const f16* Vp = Vh + headBase;

  // Kick off async staging of the first K/V tile, then stage Q transposed.
  stage_kv_async(Kp, Vp, 0, Kt[0], Vt[0], tid);
  for (int i = tid; i < 64 * 64; i += 128) {
    int ql = i >> 6, d = i & 63;
    qT[d * 64 + ql] = Qp[(size_t)(q0blk + ql) * HEADD + d];
  }
  wait_async0();
  __syncthreads();

  // Loop-invariant Q^T B-fragments for this wave (kd-steps 0 and 32).
  const int qw = w * 16;
  const v16h bq0 = load_b32x16(qT, 64, 0,  qw);
  const v16h bq1 = load_b32x16(qT, 64, 32, qw);

  float m = -1e30f, ssum = 0.f;
  v8f o0 = {}, o1 = {}, o2 = {}, o3 = {};
  const float scale = 0.125f;  // 1/sqrt(64)

  for (int kb = 0; kb < SEQ; kb += 32) {
    const int cur = (kb >> 5) & 1;
    // Prefetch next tile into the other buffer while we compute this one.
    if (kb + 32 < SEQ)
      stage_kv_async(Kp, Vp, kb + 32, Kt[cur ^ 1], Vt[cur ^ 1], tid);

    const f16* Ktc = Kt[cur];
    const f16* Vtc = Vt[cur];

    // S^T (keys x q): two key m-tiles, two kd-steps each.
    v8f st0 = {}, st1 = {};
    {
      v16h a00 = load_a16x32(Ktc, 64, 0, 0);
      v16h a01 = load_a16x32(Ktc, 64, 0, 32);
      st0 = wmma_f16(a00, bq0, st0);
      st0 = wmma_f16(a01, bq1, st0);
      v16h a10 = load_a16x32(Ktc, 64, 16, 0);
      v16h a11 = load_a16x32(Ktc, 64, 16, 32);
      st1 = wmma_f16(a10, bq0, st1);
      st1 = wmma_f16(a11, bq1, st1);
    }

    // Scale + tile max (per q = lane%16; lanes l and l+16 hold the two
    // key half-rows of the same q column -> combine with shfl_xor(16)).
    float tmax = -1e30f;
#pragma unroll
    for (int r = 0; r < 8; ++r) {
      st0[r] *= scale; st1[r] *= scale;
      tmax = fmaxf(tmax, fmaxf(st0[r], st1[r]));
    }
    tmax = fmaxf(tmax, __shfl_xor(tmax, 16, 32));
    const float mnew  = fmaxf(m, tmax);
    const float alpha = __expf(m - mnew);

    float rsum = 0.f;
#pragma unroll
    for (int r = 0; r < 8; ++r) {
      st0[r] = __expf(st0[r] - mnew);
      st1[r] = __expf(st1[r] - mnew);
      rsum += st0[r] + st1[r];
    }
    rsum += __shfl_xor(rsum, 16, 32);
    ssum = ssum * alpha + rsum;
    m = mnew;

    // Write P^T back transposed into per-wave LDS as [q][key].
    // C-layout: VGPR r, lanes<16 -> (key=r, q=lane); lanes>=16 -> (key=8+r).
    f16* pb = pBuf[w];
    const int qn      = l & 15;
    const int keyBase = (l >> 4) * 8;
#pragma unroll
    for (int r = 0; r < 8; ++r) {
      pb[qn * 32 + keyBase + r]        = (f16)st0[r];
      pb[qn * 32 + 16 + keyBase + r]   = (f16)st1[r];
    }
    asm volatile("s_wait_dscnt 0x0" ::: "memory");  // order P write -> A-frag read

    // Rescale running O by alpha(q); q = r (lanes<16) or 8+r (lanes>=16).
    float arf[8];
#pragma unroll
    for (int r = 0; r < 8; ++r) {
      float alo = __shfl(alpha, r, 32);
      float ahi = __shfl(alpha, 8 + r, 32);
      arf[r] = (l < 16) ? alo : ahi;
    }
#pragma unroll
    for (int r = 0; r < 8; ++r) {
      o0[r] *= arf[r]; o1[r] *= arf[r]; o2[r] *= arf[r]; o3[r] *= arf[r];
    }

    // O += P(16q x 32key) * V(32key x 64d): 4 dv tiles, one WMMA each.
    v16h ap  = load_a16x32(pb, 32, 0, 0);
    v16h bv0 = load_b32x16(Vtc, 64, 0, 0);
    v16h bv1 = load_b32x16(Vtc, 64, 0, 16);
    v16h bv2 = load_b32x16(Vtc, 64, 0, 32);
    v16h bv3 = load_b32x16(Vtc, 64, 0, 48);
    o0 = wmma_f16(ap, bv0, o0);
    o1 = wmma_f16(ap, bv1, o1);
    o2 = wmma_f16(ap, bv2, o2);
    o3 = wmma_f16(ap, bv3, o3);

    // Drain this wave's async copies of the *next* tile, then barrier so no
    // wave reads the fresh buffer (or overwrites this one) early.
    wait_async0();
    __syncthreads();
  }

  // Normalize by running sum and write concat-head output (f16).
  float inv[8];
#pragma unroll
  for (int r = 0; r < 8; ++r) {
    float slo = __shfl(ssum, r, 32);
    float shi = __shfl(ssum, 8 + r, 32);
    inv[r] = 1.0f / ((l < 16) ? slo : shi);
  }
  const int nn     = l & 15;
  const int rowAdd = (l >> 4) * 8;
#pragma unroll
  for (int r = 0; r < 8; ++r) {
    size_t row = (size_t)b * SEQ + q0blk + w * 16 + rowAdd + r;
    f16* op = Ah + row * DMODEL + h * HEADD + nn;
    op[0]  = (f16)(o0[r] * inv[r]);
    op[16] = (f16)(o1[r] * inv[r]);
    op[32] = (f16)(o2[r] * inv[r]);
    op[48] = (f16)(o3[r] * inv[r]);
  }
}

// ---------------------------------------------------------------------------
// Output projection: [8192 x 1024] x WoT[1024 x 1024] + bo -> f32 d_out.
// ---------------------------------------------------------------------------
__global__ void __launch_bounds__(256)
out_gemm(const f16* __restrict__ Ah, const f16* __restrict__ WoT,
         const float* __restrict__ bo, float* __restrict__ out) {
  const int tid = threadIdx.x;
  const int w = tid >> 5, l = tid & 31;
  const int m0 = blockIdx.x * 32 + (w >> 2) * 16;
  const int n0 = blockIdx.y * 64 + (w & 3) * 16;

  v8f c0 = {}, c1 = {};
  for (int k0 = 0; k0 < DMODEL; k0 += 64) {
    if (k0 + 256 < DMODEL)
      __builtin_prefetch(Ah + (size_t)(m0 + (l & 15)) * DMODEL + k0 + 256, 0, 1);
    v16h a0 = load_a16x32(Ah, DMODEL, m0, k0);
    v16h b0 = load_b32x16(WoT, DMODEL, k0, n0);
    c0 = wmma_f16(a0, b0, c0);
    v16h a1 = load_a16x32(Ah, DMODEL, m0, k0 + 32);
    v16h b1 = load_b32x16(WoT, DMODEL, k0 + 32, n0);
    c1 = wmma_f16(a1, b1, c1);
  }
  v8f c = c0 + c1;

  const float bb     = bo[n0 + (l & 15)];
  const int   rowAdd = (l >> 4) * 8;
#pragma unroll
  for (int r = 0; r < 8; ++r) {
    size_t grow = (size_t)(m0 + rowAdd + r);
    out[grow * DMODEL + n0 + (l & 15)] = c[r] + bb;
  }
}

// ---------------------------------------------------------------------------
// Host launcher. Workspace layout (bytes):
//   0      xh   [8192x1024] f16   (16 MiB)
//   16M    Wqh  [16x1024x64] f16  ( 2 MiB)
//   18M    Wkh                      2 MiB
//   20M    Wvh                      2 MiB
//   22M    WoT  [1024x1024] f16    2 MiB
//   24M    Qh   [B,H,S,64] f16    16 MiB
//   40M    Kh                     16 MiB
//   56M    Vh                     16 MiB
//   72M    Ah   [8192x1024] f16   16 MiB   (total 88 MiB)
// ---------------------------------------------------------------------------
extern "C" void kernel_launch(void* const* d_in, const int* in_sizes, int n_in,
                              void* d_out, int out_size, void* d_ws, size_t ws_size,
                              hipStream_t stream) {
  (void)in_sizes; (void)n_in; (void)out_size; (void)ws_size;
  const float* x  = (const float*)d_in[0];
  const float* Wq = (const float*)d_in[1];
  const float* bq = (const float*)d_in[2];
  const float* Wk = (const float*)d_in[3];
  const float* bk = (const float*)d_in[4];
  const float* Wv = (const float*)d_in[5];
  const float* bv = (const float*)d_in[6];
  const float* Wo = (const float*)d_in[7];
  const float* bo = (const float*)d_in[8];

  char* ws = (char*)d_ws;
  const size_t MB = 1u << 20;
  f16* xh  = (f16*)(ws);
  f16* Wqh = (f16*)(ws + 16 * MB);
  f16* Wkh = (f16*)(ws + 18 * MB);
  f16* Wvh = (f16*)(ws + 20 * MB);
  f16* WoT = (f16*)(ws + 22 * MB);
  f16* Qh  = (f16*)(ws + 24 * MB);
  f16* Kh  = (f16*)(ws + 40 * MB);
  f16* Vh  = (f16*)(ws + 56 * MB);
  f16* Ah  = (f16*)(ws + 72 * MB);

  const int nx = BS * DMODEL;              // 8,388,608
  const int nw = NHEADS * DMODEL * HEADD;  // 1,048,576

  cvt_f32_f16<<<(nx + 255) / 256, 256, 0, stream>>>(x, xh, nx);
  cvt_f32_f16<<<(nw + 255) / 256, 256, 0, stream>>>(Wq, Wqh, nw);
  cvt_f32_f16<<<(nw + 255) / 256, 256, 0, stream>>>(Wk, Wkh, nw);
  cvt_f32_f16<<<(nw + 255) / 256, 256, 0, stream>>>(Wv, Wvh, nw);
  cvt_wo_transpose<<<(DMODEL * DMODEL) / 256, 256, 0, stream>>>(Wo, WoT);

  qkv_gemm<<<dim3(BS / 32, NHEADS, 3), 256, 0, stream>>>(
      xh, Wqh, Wkh, Wvh, bq, bk, bv, Qh, Kh, Vh);

  flash_attn<<<dim3(SEQ / 64, NHEADS, BATCH), 128, 0, stream>>>(Qh, Kh, Vh, Ah);

  out_gemm<<<dim3(BS / 32, DMODEL / 64), 256, 0, stream>>>(Ah, WoT, bo, (float*)d_out);
}
